// MambaDecoderLayer_19404662243340
// MI455X (gfx1250) — matrix operand order repaired and checked by reading, not compiled, not run on hardware
//
#include <hip/hip_runtime.h>
#include <hip/hip_bf16.h>
#include <math.h>

// ---------------------------------------------------------------------------
// Types for CDNA5 WMMA (wave32, 16x16x32 bf16 -> f32)
// ---------------------------------------------------------------------------
typedef __bf16 bf16_t;
typedef __attribute__((ext_vector_type(8)))  __bf16 v8bf;
typedef __attribute__((ext_vector_type(16))) __bf16 v16bf;
typedef __attribute__((ext_vector_type(8)))  float  v8f;

#define D_MODEL 512
#define D_INNER 1024
#define NHEADS  16
#define HEADDIM 64
#define DSTATE  64
#define DCONV   4
#define DFF     2048
#define NB      16
#define LM      1024
#define LT      512
#define LTOT    1536            // LM + LT
#define CONVDIM 1152            // D_INNER + 2*DSTATE
#define DINPROJ 2192            // 2*D_INNER + 2*DSTATE + NHEADS

// ---------------------------------------------------------------------------
// Helpers
// ---------------------------------------------------------------------------
__device__ __forceinline__ bf16_t f2bf(float f) {
  union { float f; unsigned u; } c; c.f = f;
  unsigned r = (c.u + 0x7FFFu + ((c.u >> 16) & 1u)) >> 16;
  union { unsigned short s; bf16_t b; } o; o.s = (unsigned short)r;
  return o.b;
}

__device__ __forceinline__ float siluf(float x) { return x / (1.f + __expf(-x)); }

__device__ __forceinline__ float blockReduceSum256(float v) {
  __shared__ float sm[256];
  sm[threadIdx.x] = v;
  __syncthreads();
  for (int s = 128; s > 0; s >>= 1) {
    if ((int)threadIdx.x < s) sm[threadIdx.x] += sm[threadIdx.x + s];
    __syncthreads();
  }
  return sm[0];
}

// ---------------------------------------------------------------------------
// WMMA fragment loads (contiguous k order, same for A and B)
// A-matrix 16x32 bf16: lane l holds row m=l%16; half=l/16:
//   elems 0..15 -> k = half*16 + 0..15
// B-matrix 32x16 bf16: lane l holds col n=l%16; half=l/16:
//   elems 0..15 -> k = half*16 + 0..15   (one contiguous run of W[n][k])
// ---------------------------------------------------------------------------
__device__ __forceinline__ v16bf load_a_frag(const bf16_t* __restrict__ A,
                                             int lda, int m0, int kb, int M) {
  int lane = threadIdx.x & 31;
  int m = m0 + (lane & 15); if (m >= M) m = M - 1;
  int half = lane >> 4;
  const bf16_t* p = A + (size_t)m * lda + kb + half * 16;
  v8bf lo = *(const v8bf*)(p);
  v8bf hi = *(const v8bf*)(p + 8);
  return __builtin_shufflevector(lo, hi, 0,1,2,3,4,5,6,7,8,9,10,11,12,13,14,15);
}

__device__ __forceinline__ v16bf load_b_frag(const bf16_t* __restrict__ W,
                                             int ldb, int n0, int kb, int N) {
  int lane = threadIdx.x & 31;
  int n = n0 + (lane & 15); if (n >= N) n = N - 1;
  int half = lane >> 4;
  return *(const v16bf*)(W + (size_t)n * ldb + kb + half * 16);
}

// ---------------------------------------------------------------------------
// Generic WMMA bf16 GEMM:  C(MxN) = act(A(MxK) * W(NxK)^T + bias) + res
//   act: 0 = none, 1 = exact GELU
//   Cf (f32) and/or Cb (bf16) outputs; whichever is non-null is written.
// Workgroup = 8 waves as 2(M) x 4(N); wave tile = 32x64 (2 A frags x 4 B
// frags = 8 WMMAs per K-step, A reused 4x, B reused 2x). Block tile 64x256.
// ---------------------------------------------------------------------------
__global__ __launch_bounds__(256) void gemm_bf16_wmma(
    const bf16_t* __restrict__ A, int lda,
    const bf16_t* __restrict__ W, int ldb,
    float* __restrict__ Cf, bf16_t* __restrict__ Cb, int ldc,
    const float* __restrict__ bias,
    const float* __restrict__ res, int ldr,
    int M, int N, int K, int act)
{
  int wave = threadIdx.x >> 5;
  int wr = wave >> 2;        // 0..1  (M)
  int wc = wave & 3;         // 0..3  (N)
  int m0 = blockIdx.y * 64 + wr * 32;
  int n0 = blockIdx.x * 256 + wc * 64;
  if (m0 >= M || n0 >= N) return;   // wave-uniform

  v8f acc[2][4];
#pragma unroll
  for (int mi = 0; mi < 2; ++mi)
#pragma unroll
    for (int ni = 0; ni < 4; ++ni) acc[mi][ni] = (v8f){};

  for (int kb = 0; kb < K; kb += 32) {
    v16bf a0 = load_a_frag(A, lda, m0,      kb, M);
    v16bf a1 = load_a_frag(A, lda, m0 + 16, kb, M);
    v16bf b[4];
#pragma unroll
    for (int ni = 0; ni < 4; ++ni)
      b[ni] = load_b_frag(W, ldb, n0 + 16 * ni, kb, N);

    if (kb + 32 < K) {
      int lane = threadIdx.x & 31;
      int n = n0 + (lane & 15); if (n >= N) n = N - 1;
      __builtin_prefetch(W + (size_t)n * ldb + kb + 32, 0, 1);
    }

#pragma unroll
    for (int ni = 0; ni < 4; ++ni) {
      acc[0][ni] = __builtin_amdgcn_wmma_f32_16x16x32_bf16(
          false, a0, false, b[ni], (short)0, acc[0][ni], false, false);
      acc[1][ni] = __builtin_amdgcn_wmma_f32_16x16x32_bf16(
          false, a1, false, b[ni], (short)0, acc[1][ni], false, false);
    }
  }

  int lane = threadIdx.x & 31;
#pragma unroll
  for (int mi = 0; mi < 2; ++mi) {
#pragma unroll
    for (int ni = 0; ni < 4; ++ni) {
      int n = n0 + 16 * ni + (lane & 15);
      int mbase = m0 + 16 * mi + (lane >> 4) * 8;
      float bv = (bias != nullptr && n < N) ? bias[n] : 0.f;
#pragma unroll
      for (int j = 0; j < 8; ++j) {
        int m = mbase + j;
        if (m < M && n < N) {
          float x = acc[mi][ni][j] + bv;
          if (act == 1) x = 0.5f * x * (1.f + erff(x * 0.70710678118654752f));
          if (res != nullptr) x += res[(size_t)m * ldr + n];
          if (Cf != nullptr) Cf[(size_t)m * ldc + n] = x;
          if (Cb != nullptr) Cb[(size_t)m * ldc + n] = f2bf(x);
        }
      }
    }
  }
}

// ---------------------------------------------------------------------------
// fp32 -> bf16 weight conversion
// ---------------------------------------------------------------------------
__global__ void f32_to_bf16(const float* __restrict__ in,
                            bf16_t* __restrict__ out, int n) {
  int i = blockIdx.x * 256 + threadIdx.x;
  if (i < n) out[i] = f2bf(in[i]);
}

// ---------------------------------------------------------------------------
// Build u = concat(memory*mask^2, rmsnorm(tgt*mask)*mask) as bf16.
// Also stash masked tgt (f32) for the residual.
// One block per sequence row.
// ---------------------------------------------------------------------------
__global__ __launch_bounds__(256) void prep_kernel(
    const float* __restrict__ tgt, const float* __restrict__ memory,
    const float* __restrict__ mem_mask, const float* __restrict__ tgt_mask,
    bf16_t* __restrict__ u_bf, float* __restrict__ tgt_masked)
{
  int row = blockIdx.x;                 // 0 .. NB*LTOT-1
  int b = row / LTOT, l = row % LTOT;
  int tid = threadIdx.x;
  if (l < LM) {
    float mk = mem_mask[b * LM + l];
    float mk2 = mk * mk;                // memory masked twice in reference
    const float* src = memory + ((size_t)b * LM + l) * D_MODEL;
    bf16_t* dst = u_bf + (size_t)row * D_MODEL;
    for (int d = tid; d < D_MODEL; d += 256) dst[d] = f2bf(src[d] * mk2);
  } else {
    int t = l - LM;
    float mk = tgt_mask[b * LT + t];
    const float* src = tgt + ((size_t)b * LT + t) * D_MODEL;
    float* tm = tgt_masked + ((size_t)b * LT + t) * D_MODEL;
    float v[2]; float ss = 0.f;
#pragma unroll
    for (int i = 0; i < 2; ++i) {
      int d = tid + i * 256;
      float x = src[d] * mk;
      v[i] = x; tm[d] = x; ss += x * x;
    }
    ss = blockReduceSum256(ss);
    __shared__ float s_inv;
    if (tid == 0) s_inv = rsqrtf(ss / (float)D_MODEL + 1e-6f);
    __syncthreads();
    float inv = s_inv;
    bf16_t* dst = u_bf + (size_t)row * D_MODEL;
#pragma unroll
    for (int i = 0; i < 2; ++i) {
      int d = tid + i * 256;
      dst[d] = f2bf(v[i] * inv * mk);
    }
  }
}

// ---------------------------------------------------------------------------
// Depthwise causal conv (DCONV=4) + SiLU on xBC, plus dt softplus / a=exp(dt*A)
// One block per (b,l) row of zxbcdt.
// ---------------------------------------------------------------------------
__global__ __launch_bounds__(256) void conv_dt_kernel(
    const float* __restrict__ zx, const float* __restrict__ conv_w,
    const float* __restrict__ conv_b, const float* __restrict__ dt_bias,
    const float* __restrict__ A_log,
    float* __restrict__ xbuf, float* __restrict__ Bbuf, float* __restrict__ Cbuf,
    float* __restrict__ dtbuf, float* __restrict__ abuf)
{
  int row = blockIdx.x;
  int b = row / LTOT, l = row % LTOT;
  const float* zr = zx + (size_t)row * DINPROJ;
  for (int c = threadIdx.x; c < CONVDIM + NHEADS; c += 256) {
    if (c < CONVDIM) {
      float acc = conv_b[c];
#pragma unroll
      for (int k = 0; k < DCONV; ++k) {
        int ll = l - (DCONV - 1) + k;
        if (ll >= 0)
          acc += conv_w[c * DCONV + k] *
                 zx[((size_t)b * LTOT + ll) * DINPROJ + D_INNER + c];
      }
      float s = siluf(acc);
      if (c < D_INNER)            xbuf[(size_t)row * D_INNER + c] = s;
      else if (c < D_INNER + DSTATE) Bbuf[(size_t)row * DSTATE + (c - D_INNER)] = s;
      else                        Cbuf[(size_t)row * DSTATE + (c - D_INNER - DSTATE)] = s;
    } else {
      int h = c - CONVDIM;
      float d = zr[2 * D_INNER + 2 * DSTATE + h] + dt_bias[h];
      float sp = (d > 20.f) ? d : log1pf(__expf(d));
      dtbuf[(size_t)row * NHEADS + h] = sp;
      abuf[(size_t)row * NHEADS + h] = __expf(sp * (-__expf(A_log[h])));
    }
  }
}

// ---------------------------------------------------------------------------
// Sequential selective-state scan. One block per (b,h); 64 threads, thread p
// holds state row h[p][0..63] in 64 VGPRs. y written only for l >= LM.
// ---------------------------------------------------------------------------
__global__ __launch_bounds__(64) void scan_kernel(
    const float* __restrict__ xbuf, const float* __restrict__ Bbuf,
    const float* __restrict__ Cbuf, const float* __restrict__ dtbuf,
    const float* __restrict__ abuf, const float* __restrict__ Dvec,
    float* __restrict__ ybuf)
{
  int bh = blockIdx.x;
  int b = bh >> 4, h = bh & 15;
  int p = threadIdx.x;
  float Dh = Dvec[h];
  float hreg[DSTATE];
#pragma unroll
  for (int n = 0; n < DSTATE; ++n) hreg[n] = 0.f;

  __shared__ float Bs[DSTATE], Cs[DSTATE], xs[HEADDIM];

  for (int l = 0; l < LTOT; ++l) {
    size_t row = (size_t)b * LTOT + l;
    __syncthreads();
    Bs[p] = Bbuf[row * DSTATE + p];
    Cs[p] = Cbuf[row * DSTATE + p];
    xs[p] = xbuf[(row * NHEADS + h) * HEADDIM + p];
    float a_t  = abuf[row * NHEADS + h];
    float dt_t = dtbuf[row * NHEADS + h];
    __syncthreads();
    float dx = dt_t * xs[p];
    float acc = 0.f;
#pragma unroll
    for (int n = 0; n < DSTATE; ++n) {
      hreg[n] = fmaf(a_t, hreg[n], dx * Bs[n]);
      acc = fmaf(hreg[n], Cs[n], acc);
    }
    if (l >= LM) {
      size_t orow = (size_t)b * LT + (l - LM);
      ybuf[(orow * NHEADS + h) * HEADDIM + p] = acc + Dh * xs[p];
    }
  }
}

// ---------------------------------------------------------------------------
// y = rmsnorm(y * silu(z), 1e-5) * norm_w -> bf16. One block per output row.
// ---------------------------------------------------------------------------
__global__ __launch_bounds__(256) void gate_norm_kernel(
    const float* __restrict__ ybuf, const float* __restrict__ zx,
    const float* __restrict__ norm_w, bf16_t* __restrict__ ybf)
{
  int r = blockIdx.x;                    // 0 .. NB*LT-1
  int b = r / LT, t = r % LT;
  size_t zrow = ((size_t)b * LTOT + LM + t) * DINPROJ;
  const float* y = ybuf + (size_t)r * D_INNER;
  float v[4]; float ss = 0.f;
#pragma unroll
  for (int i = 0; i < 4; ++i) {
    int c = threadIdx.x + i * 256;
    float z = zx[zrow + c];
    float x = y[c] * siluf(z);
    v[i] = x; ss += x * x;
  }
  ss = blockReduceSum256(ss);
  __shared__ float s_inv;
  if (threadIdx.x == 0) s_inv = rsqrtf(ss / (float)D_INNER + 1e-5f);
  __syncthreads();
  float inv = s_inv;
#pragma unroll
  for (int i = 0; i < 4; ++i) {
    int c = threadIdx.x + i * 256;
    ybf[(size_t)r * D_INNER + c] = f2bf(v[i] * inv * norm_w[c]);
  }
}

// ---------------------------------------------------------------------------
// rmsnorm(x, 1e-6) over D_MODEL -> bf16 (pre-FFN norm)
// ---------------------------------------------------------------------------
__global__ __launch_bounds__(256) void rmsnorm512_kernel(
    const float* __restrict__ in, bf16_t* __restrict__ out)
{
  int r = blockIdx.x;
  const float* x = in + (size_t)r * D_MODEL;
  float v[2]; float ss = 0.f;
#pragma unroll
  for (int i = 0; i < 2; ++i) {
    int d = threadIdx.x + i * 256;
    float t = x[d]; v[i] = t; ss += t * t;
  }
  ss = blockReduceSum256(ss);
  __shared__ float s_inv;
  if (threadIdx.x == 0) s_inv = rsqrtf(ss / (float)D_MODEL + 1e-6f);
  __syncthreads();
  float inv = s_inv;
#pragma unroll
  for (int i = 0; i < 2; ++i) {
    int d = threadIdx.x + i * 256;
    out[(size_t)r * D_MODEL + d] = f2bf(v[i] * inv);
  }
}

// ---------------------------------------------------------------------------
// Host launcher
// ---------------------------------------------------------------------------
static inline size_t alignup(size_t x) { return (x + 255) & ~(size_t)255; }

extern "C" void kernel_launch(void* const* d_in, const int* in_sizes, int n_in,
                              void* d_out, int out_size, void* d_ws, size_t ws_size,
                              hipStream_t stream) {
  (void)in_sizes; (void)n_in; (void)out_size; (void)ws_size;
  const float* tgt        = (const float*)d_in[0];
  const float* memory     = (const float*)d_in[1];
  const float* mem_mask   = (const float*)d_in[2];
  const float* tgt_mask   = (const float*)d_in[3];
  const float* in_proj_w  = (const float*)d_in[4];
  const float* conv_w     = (const float*)d_in[5];
  const float* conv_b     = (const float*)d_in[6];
  const float* dt_bias    = (const float*)d_in[7];
  const float* A_log      = (const float*)d_in[8];
  const float* Dvec       = (const float*)d_in[9];
  const float* norm_w     = (const float*)d_in[10];
  const float* out_proj_w = (const float*)d_in[11];
  const float* fc1_w      = (const float*)d_in[12];
  const float* fc1_b      = (const float*)d_in[13];
  const float* fc2_w      = (const float*)d_in[14];
  const float* fc2_b      = (const float*)d_in[15];

  const int Mfull = NB * LTOT;          // 24576
  const int Mtail = NB * LT;            // 8192

  char* w = (char*)d_ws;
  size_t off = 0;
  auto take = [&](size_t bytes) { char* p = w + off; off += alignup(bytes); return p; };

  bf16_t* wb_inproj  = (bf16_t*)take((size_t)DINPROJ * D_MODEL * 2);
  bf16_t* wb_outproj = (bf16_t*)take((size_t)D_MODEL * D_INNER * 2);
  bf16_t* wb_fc1     = (bf16_t*)take((size_t)DFF * D_MODEL * 2);
  bf16_t* wb_fc2     = (bf16_t*)take((size_t)D_MODEL * DFF * 2);
  bf16_t* u_bf       = (bf16_t*)take((size_t)Mfull * D_MODEL * 2);
  float*  tgt_masked = (float*) take((size_t)Mtail * D_MODEL * 4);
  float*  zx         = (float*) take((size_t)Mfull * DINPROJ * 4);
  float*  xbuf       = (float*) take((size_t)Mfull * D_INNER * 4);
  float*  Bbuf       = (float*) take((size_t)Mfull * DSTATE * 4);
  float*  Cbuf       = (float*) take((size_t)Mfull * DSTATE * 4);
  float*  dtbuf      = (float*) take((size_t)Mfull * NHEADS * 4);
  float*  abuf       = (float*) take((size_t)Mfull * NHEADS * 4);
  float*  ybuf       = (float*) take((size_t)Mtail * D_INNER * 4);
  bf16_t* ybf        = (bf16_t*)take((size_t)Mtail * D_INNER * 2);
  float*  tgtres     = (float*) take((size_t)Mtail * D_MODEL * 4);
  bf16_t* t1bf       = (bf16_t*)take((size_t)Mtail * D_MODEL * 2);
  bf16_t* hbf        = (bf16_t*)take((size_t)Mtail * DFF * 2);

  // 0) weights to bf16
  {
    int n;
    n = DINPROJ * D_MODEL; f32_to_bf16<<<(n + 255) / 256, 256, 0, stream>>>(in_proj_w, wb_inproj, n);
    n = D_MODEL * D_INNER; f32_to_bf16<<<(n + 255) / 256, 256, 0, stream>>>(out_proj_w, wb_outproj, n);
    n = DFF * D_MODEL;     f32_to_bf16<<<(n + 255) / 256, 256, 0, stream>>>(fc1_w, wb_fc1, n);
    n = D_MODEL * DFF;     f32_to_bf16<<<(n + 255) / 256, 256, 0, stream>>>(fc2_w, wb_fc2, n);
  }

  // 1) masks + pre-norm, build u (bf16) and masked tgt residual
  prep_kernel<<<Mfull, 256, 0, stream>>>(tgt, memory, mem_mask, tgt_mask, u_bf, tgt_masked);

  // 2) zxbcdt = u @ in_proj_w^T   (24576 x 2192, K=512)
  {
    dim3 grid((DINPROJ + 255) / 256, (Mfull + 63) / 64);
    gemm_bf16_wmma<<<grid, 256, 0, stream>>>(u_bf, D_MODEL, wb_inproj, D_MODEL,
        zx, nullptr, DINPROJ, nullptr, nullptr, 0, Mfull, DINPROJ, D_MODEL, 0);
  }

  // 3) conv + silu + dt/a
  conv_dt_kernel<<<Mfull, 256, 0, stream>>>(zx, conv_w, conv_b, dt_bias, A_log,
                                            xbuf, Bbuf, Cbuf, dtbuf, abuf);

  // 4) sequential scan (+ D*x), keep only last LT rows
  scan_kernel<<<NB * NHEADS, 64, 0, stream>>>(xbuf, Bbuf, Cbuf, dtbuf, abuf, Dvec, ybuf);

  // 5) gate + rmsnorm(1e-5)*norm_w -> bf16
  gate_norm_kernel<<<Mtail, 256, 0, stream>>>(ybuf, zx, norm_w, ybf);

  // 6) tgtres = tgt_masked + ybf @ out_proj_w^T   (8192 x 512, K=1024)
  {
    dim3 grid((D_MODEL + 255) / 256, (Mtail + 63) / 64);
    gemm_bf16_wmma<<<grid, 256, 0, stream>>>(ybf, D_INNER, wb_outproj, D_INNER,
        tgtres, nullptr, D_MODEL, nullptr, tgt_masked, D_MODEL,
        Mtail, D_MODEL, D_INNER, 0);
  }

  // 7) pre-FFN rmsnorm(1e-6) -> bf16
  rmsnorm512_kernel<<<Mtail, 256, 0, stream>>>(tgtres, t1bf);

  // 8) h = gelu(t1 @ fc1^T + b1) -> bf16   (8192 x 2048, K=512)
  {
    dim3 grid((DFF + 255) / 256, (Mtail + 63) / 64);
    gemm_bf16_wmma<<<grid, 256, 0, stream>>>(t1bf, D_MODEL, wb_fc1, D_MODEL,
        nullptr, hbf, DFF, fc1_b, nullptr, 0, Mtail, DFF, D_MODEL, 1);
  }

  // 9) out = tgtres + h @ fc2^T + b2 -> d_out (f32)   (8192 x 512, K=2048)
  {
    dim3 grid((D_MODEL + 255) / 256, (Mtail + 63) / 64);
    gemm_bf16_wmma<<<grid, 256, 0, stream>>>(hbf, DFF, wb_fc2, DFF,
        (float*)d_out, nullptr, D_MODEL, fc2_b, tgtres, D_MODEL,
        Mtail, D_MODEL, DFF, 0);
  }
}
